// AxialSelfAttentionModule_2482491097697
// MI455X (gfx1250) — compile-verified, matrix operand order checked by reading, not run on hardware
//
#include <hip/hip_runtime.h>
#include <hip/hip_bf16.h>
#include <math.h>

typedef __attribute__((ext_vector_type(16))) __bf16 v16bf;
typedef __attribute__((ext_vector_type(8)))  float  v8f;

union Frag16 { v16bf v; unsigned short h[16]; };

__device__ __forceinline__ unsigned short f2bf(float f){
  union { float f; unsigned u; } x; x.f = f;
  unsigned r = x.u + 0x7FFFu + ((x.u >> 16) & 1u);
  return (unsigned short)(r >> 16);
}
__device__ __forceinline__ float bf2f(unsigned short h){
  union { unsigned u; float f; } x; x.u = ((unsigned)h) << 16; return x.f;
}

__device__ __forceinline__ v8f wmma_bf16(v16bf a, v16bf b, v8f c){
  return __builtin_amdgcn_wmma_f32_16x16x32_bf16(false, a, false, b, (short)0, c, false, false);
}

// ---- CDNA5 async global->LDS (ASYNCcnt path, bypasses VGPR staging) ----
__device__ __forceinline__ unsigned lds_off_u32(const void* p){
  // generic (flat) LDS pointer: low 32 bits are the wave-relative LDS byte offset
  return (unsigned)(unsigned long long)p;
}
__device__ __forceinline__ void async_copy_b128(unsigned lds_byte_off, const void* gbase, unsigned gbyte_off){
  asm volatile("global_load_async_to_lds_b128 %0, %1, %2"
               :: "v"(lds_byte_off), "v"(gbyte_off),
                  "s"((unsigned long long)(size_t)gbase)
               : "memory");
}
__device__ __forceinline__ void wait_async(){
  asm volatile("s_wait_asynccnt 0" ::: "memory");
}

// A fragment (16xK tile, 16-bit): lanes 0-15 M=0..15 K={0..7,16..23}; lanes 16-31 M=0..15 K={8..15,24..31}
__device__ __forceinline__ v16bf ldsA(const unsigned short* mat, int rs, int m0, int kbase){
  int lane = threadIdx.x & 31;
  int m = m0 + (lane & 15);
  int kb = kbase + ((lane & 16) ? 8 : 0);
  const unsigned short* p = mat + m * rs + kb;
  Frag16 f;
#pragma unroll
  for (int j = 0; j < 8; ++j){ f.h[j] = p[j]; f.h[8 + j] = p[16 + j]; }
  return f.v;
}
// fragment-packed B operand: 16 contiguous halves per lane (works for LDS and global)
__device__ __forceinline__ v16bf gfrag(const unsigned short* p){
  p += (size_t)(threadIdx.x & 31) * 16;
  Frag16 f;
#pragma unroll
  for (int j = 0; j < 16; ++j) f.h[j] = p[j];
  return f.v;
}
// map B-tile element (k in [0,32), n_local in [0,16)) -> (lane, half j) of packed fragment
__device__ __forceinline__ void bpos(int k, int n_local, int& lane, int& j){
  lane = n_local + (((k >> 3) & 1) ? 16 : 0);
  j = (k & 16) ? (8 + (k & 7)) : (k & 7);
}

// ---------------- pack kernels ----------------
__global__ __launch_bounds__(256) void pack_pe_kernel(const float* __restrict__ src, unsigned short* __restrict__ dst){
  long i = (long)blockIdx.x * 256 + threadIdx.x;           // 65536*128 elements
  long tok = i >> 7; int c = (int)(i & 127);
  int b = (int)(tok >> 15); int sp = (int)(tok & 32767);
  dst[i] = f2bf(src[((long)b * 128 + c) * 32768 + sp]);
}

// generic [N][128] f32 -> fragment-major bf16 pack, NT = N/16 column tiles, K=128 (4 k-tiles)
__global__ __launch_bounds__(256) void pack_wtok_kernel(const float* __restrict__ w, int NT, unsigned short* __restrict__ dst){
  int i = blockIdx.x * 256 + threadIdx.x;                  // 4*NT*512 total
  int j = i & 15; int lane = (i >> 4) & 31;
  int nt = (i >> 9) % NT; int kt = (i >> 9) / NT;
  int n = nt * 16 + (lane & 15);
  int kb = (lane & 16) ? 8 : 0;
  int k = kt * 32 + ((j < 8) ? (kb + j) : (kb + 16 + (j - 8)));
  dst[i] = f2bf(w[n * 128 + k]);
}

// lp_w1 (co,ci,3,3,3) -> [tap][ktile][ntile][lane][16]
__global__ __launch_bounds__(256) void pack_wconv_kernel(const float* __restrict__ w, unsigned short* __restrict__ dst){
  int i = blockIdx.x * 256 + threadIdx.x;                  // 27*4*8*512 = 442368
  int j = i & 15; int lane = (i >> 4) & 31;
  int nt = (i >> 9) & 7; int kt = (i >> 12) & 3; int tap = i >> 14;
  int co = nt * 16 + (lane & 15);
  int kb = (lane & 16) ? 8 : 0;
  int ci = kt * 32 + ((j < 8) ? (kb + j) : (kb + 16 + (j - 8)));
  dst[i] = f2bf(w[(co * 128 + ci) * 27 + tap]);
}

// ---------------- rotation matrices ----------------
__device__ __forceinline__ float dot3(const float* a, const float* b){ return a[0]*b[0]+a[1]*b[1]+a[2]*b[2]; }
__device__ __forceinline__ void cross3(const float* a, const float* b, float* c){
  c[0]=a[1]*b[2]-a[2]*b[1]; c[1]=a[2]*b[0]-a[0]*b[2]; c[2]=a[0]*b[1]-a[1]*b[0];
}

__global__ void rot_kernel(const float* Ad, const float* Ah, const float* Aw,
                           const float* Rd, const float* Rh, const float* Rw, float* rot){
  int t = threadIdx.x; if (t >= 24) return;
  int axis = t >> 3, h = t & 7;
  const float* A6 = (axis == 0 ? Ad : (axis == 1 ? Ah : Aw)) + h * 6;
  const float* Ri = (axis == 0 ? Rd : (axis == 1 ? Rh : Rw)) + h * 9;
  float v1[3], v2[3], v3[3], tmp[3];
  // _r6_to_matrix
  for (int i = 0; i < 3; ++i) v1[i] = A6[i];
  float n = sqrtf(dot3(v1, v1)) + 1e-7f;
  for (int i = 0; i < 3; ++i) v1[i] /= n;
  for (int i = 0; i < 3; ++i) v2[i] = A6[3 + i];
  float d = dot3(v2, v1);
  for (int i = 0; i < 3; ++i) v2[i] -= d * v1[i];
  n = sqrtf(dot3(v2, v2)) + 1e-7f;
  for (int i = 0; i < 3; ++i) v2[i] /= n;
  cross3(v1, v2, v3);
  cross3(v2, v3, tmp);
  if (dot3(v1, tmp) < 0.f) for (int i = 0; i < 3; ++i) v3[i] = -v3[i];
  float* oq = &rot[((axis * 8 + h) * 2 + 0) * 9];
  for (int i = 0; i < 3; ++i){ oq[i] = v1[i]; oq[3 + i] = v2[i]; oq[6 + i] = v3[i]; }
  // _ensure_matrix
  float w1[3], w2[3], w3[3];
  for (int i = 0; i < 3; ++i) w1[i] = Ri[i] + 1e-6f;
  n = fmaxf(sqrtf(dot3(w1, w1)), 1e-12f);
  for (int i = 0; i < 3; ++i) w1[i] /= n;
  for (int i = 0; i < 3; ++i) w2[i] = Ri[3 + i];
  d = dot3(w2, w1);
  for (int i = 0; i < 3; ++i) w2[i] = w2[i] - d * w1[i] + 1e-6f;
  n = fmaxf(sqrtf(dot3(w2, w2)), 1e-12f);
  for (int i = 0; i < 3; ++i) w2[i] /= n;
  cross3(w1, w2, w3);
  cross3(w2, w3, tmp);
  if (dot3(w1, tmp) < 0.f) for (int i = 0; i < 3; ++i) w3[i] = -w3[i];
  float Rn[9] = { w1[0], w1[1], w1[2], w2[0], w2[1], w2[2], w3[0], w3[1], w3[2] };
  float Rt[9], T1[9], T2[9];
  for (int i = 0; i < 3; ++i) for (int j = 0; j < 3; ++j) Rt[i * 3 + j] = Rn[j * 3 + i];
  for (int i = 0; i < 3; ++i) for (int j = 0; j < 3; ++j){
    float s = 0.f; for (int k = 0; k < 3; ++k) s += Rt[i * 3 + k] * Rn[k * 3 + j]; T1[i * 3 + j] = s; }
  for (int i = 0; i < 3; ++i) for (int j = 0; j < 3; ++j){
    float s = 0.f; for (int k = 0; k < 3; ++k) s += T1[i * 3 + k] * Rt[k * 3 + j]; T2[i * 3 + j] = s; }
  float* ov = &rot[((axis * 8 + h) * 2 + 1) * 9];
  for (int i = 0; i < 9; ++i) ov[i] = 0.5f * (Rn[i] + T2[i]);
}

// ---------------- 3x3x3 circular conv as WMMA GEMM ----------------
__device__ __forceinline__ v16bf convA(const unsigned short* rowbase, int dx, int m0, int kbase){
  int lane = threadIdx.x & 31;
  int xsrc = ((m0 + (lane & 15)) + dx + 31) & 31;          // circular in x
  int kb = kbase + ((lane & 16) ? 8 : 0);
  const unsigned short* p = rowbase + xsrc * 128 + kb;
  Frag16 f;
#pragma unroll
  for (int j = 0; j < 8; ++j){ f.h[j] = p[j]; f.h[8 + j] = p[16 + j]; }
  return f.v;
}

__global__ __launch_bounds__(256) void conv3_kernel(const unsigned short* __restrict__ pe_cl,
                                                    const unsigned short* __restrict__ w1pack,
                                                    const float* __restrict__ lp_b1,
                                                    float* __restrict__ lf_raw){
  int blk = blockIdx.x;                 // (b,z,y)
  int b = blk >> 10, z = (blk >> 5) & 31, y = blk & 31;
  __shared__ unsigned short xr[9 * 32 * 128];              // 72 KB of the 320 KB WGP LDS
  // async DMA of the 9-row halo into LDS (512 x b128 chunks per row)
  {
    const unsigned lds0 = lds_off_u32(xr);
    for (int c = threadIdx.x; c < 4608; c += 256){
      int rr = c >> 9; int rem = c & 511;                  // chunk within 8 KB row
      int dz = rr / 3, dy = rr - dz * 3;
      int zs = (z + dz + 31) & 31, ys = (y + dy + 31) & 31;
      unsigned gb = (unsigned)((((b * 32 + zs) * 32 + ys)) * 8192 + rem * 16);
      async_copy_b128(lds0 + (unsigned)c * 16u, pe_cl, gb);
    }
    wait_async();
  }
  __syncthreads();
  int wave = threadIdx.x >> 5, lane = threadIdx.x & 31;
  int ntile = wave;                                        // 8 waves x 16 output channels
  v8f acc0 = {}; v8f acc1 = {};
  for (int tap = 0; tap < 27; ++tap){
    int dz = tap / 9; int rem9 = tap - dz * 9; int dy = rem9 / 3; int dx = rem9 - dy * 3;
    const unsigned short* rowbase = &xr[(dz * 3 + dy) * 32 * 128];
    if (tap < 26)                                          // stream next tap's weights ahead
      __builtin_prefetch(w1pack + (size_t)(((tap + 1) * 4) * 8 + ntile) * 512, 0, 0);
#pragma unroll
    for (int kt = 0; kt < 4; ++kt){
      v16bf a0 = convA(rowbase, dx, 0, kt * 32);
      v16bf a1 = convA(rowbase, dx, 16, kt * 32);
      v16bf bw = gfrag(w1pack + (size_t)((tap * 4 + kt) * 8 + ntile) * 512);
      acc0 = wmma_bf16(a0, bw, acc0);
      acc1 = wmma_bf16(a1, bw, acc1);
    }
  }
  int n = ntile * 16 + (lane & 15);
  float bs = lp_b1[n];
  int tok0 = ((b * 32 + z) * 32 + y) * 32;
#pragma unroll
  for (int r = 0; r < 8; ++r){
    int m = r + ((lane & 16) ? 8 : 0);
    lf_raw[(size_t)(tok0 + m) * 128 + n]      = acc0[r] + bs;
    lf_raw[(size_t)(tok0 + 16 + m) * 128 + n] = acc1[r] + bs;
  }
}

// ---------------- per-(b,c) instance-norm stats ----------------
__global__ __launch_bounds__(256) void stats_kernel(const float* __restrict__ in, float* __restrict__ st){
  int blk = blockIdx.x; int b = blk >> 7, c = blk & 127;
  __shared__ float s1[256], s2[256];
  float a = 0.f, q = 0.f;
  for (int t = threadIdx.x; t < 32768; t += 256){
    float v = in[((size_t)b * 32768 + t) * 128 + c];
    a += v; q += v * v;
  }
  s1[threadIdx.x] = a; s2[threadIdx.x] = q; __syncthreads();
  for (int o = 128; o > 0; o >>= 1){
    if (threadIdx.x < o){ s1[threadIdx.x] += s1[threadIdx.x + o]; s2[threadIdx.x] += s2[threadIdx.x + o]; }
    __syncthreads();
  }
  if (threadIdx.x == 0){
    float mu = s1[0] * (1.f / 32768.f);
    float var = s2[0] * (1.f / 32768.f) - mu * mu;
    st[blk * 2] = mu; st[blk * 2 + 1] = rsqrtf(var + 1e-5f);
  }
}

// ---------------- fused token GEMM (M=65536, N=128, K=128) ----------------
// modes: 0 f32 channels-last; 1 bf16 channels-last; 2 sigmoid * x -> bf16 cl; 3 f32 NCDHW
__global__ __launch_bounds__(256) void token_gemm_kernel(const void* __restrict__ Ain, int a_is_f32,
                                                         const float* __restrict__ stats, int do_gelu,
                                                         const unsigned short* __restrict__ wpack,
                                                         const float* __restrict__ bias, int mode,
                                                         void* __restrict__ out,
                                                         const float* __restrict__ x_src){
  __shared__ unsigned short at[128 * 128];                 // 32 KB
  long t0 = (long)blockIdx.x * 128;
  for (int i = threadIdx.x; i < 128 * 128; i += 256){
    int row = i >> 7, col = i & 127;
    long tok = t0 + row;
    float v = a_is_f32 ? ((const float*)Ain)[tok * 128 + col]
                       : bf2f(((const unsigned short*)Ain)[tok * 128 + col]);
    if (stats){
      int b = (int)(tok >> 15);
      const float* s = &stats[(b * 128 + col) * 2];
      v = (v - s[0]) * s[1];
    }
    if (do_gelu) v = 0.5f * v * (1.f + erff(v * 0.70710678f));
    at[i] = f2bf(v);
  }
  __syncthreads();
  int wave = threadIdx.x >> 5, lane = threadIdx.x & 31;
  int m0 = wave * 16;
  v8f acc[8];
#pragma unroll
  for (int i = 0; i < 8; ++i) acc[i] = (v8f){};
#pragma unroll
  for (int kt = 0; kt < 4; ++kt){
    v16bf a = ldsA(at, 128, m0, kt * 32);
#pragma unroll
    for (int nt = 0; nt < 8; ++nt){
      v16bf bw = gfrag(wpack + (size_t)(kt * 8 + nt) * 512);
      acc[nt] = wmma_bf16(a, bw, acc[nt]);
    }
  }
#pragma unroll
  for (int nt = 0; nt < 8; ++nt){
    int n = nt * 16 + (lane & 15);
    float bs = bias ? bias[n] : 0.f;
#pragma unroll
    for (int r = 0; r < 8; ++r){
      int m = m0 + r + ((lane & 16) ? 8 : 0);
      long tok = t0 + m;
      float v = acc[nt][r] + bs;
      if (mode == 0) ((float*)out)[tok * 128 + n] = v;
      else if (mode == 1) ((unsigned short*)out)[tok * 128 + n] = f2bf(v);
      else if (mode == 2){
        float s = 1.f / (1.f + __expf(-v));
        int b = (int)(tok >> 15), sp = (int)(tok & 32767);
        float xv = x_src[((long)b * 128 + n) * 32768 + sp];
        ((unsigned short*)out)[tok * 128 + n] = f2bf(xv * s);
      } else {
        int b = (int)(tok >> 15), sp = (int)(tok & 32767);
        ((float*)out)[((long)b * 128 + n) * 32768 + sp] = v;
      }
    }
  }
}

// ---------------- fused axial attention ----------------
// one workgroup (4 waves) per line of L=32 tokens
#define SM_XIN 0        // 32x128 bf16, 4096 halves
#define SM_QKV 4096     // [3][8][32][16], 12288 halves
#define SM_Q   16384    // [8][32][32] (cols 16-31 zero), A-layout, 8192 halves
#define SM_KP  24576    // [8][2][32lane][16] fragment-packed k^T, 8192 halves
#define SM_VP  32768    // [8][32lane][16] fragment-packed v, 4096 halves
#define SM_P   0        // [8][32][32] softmax probs (reuses xin/qkv)

__global__ __launch_bounds__(128) void axial_attn_kernel(const unsigned short* __restrict__ xmod,
                                                         const unsigned short* __restrict__ qkvpack,
                                                         const float* __restrict__ qkv_b,
                                                         const float* __restrict__ rotb,
                                                         const float* __restrict__ pbw,
                                                         int axis, int comp,
                                                         float* __restrict__ outacc){
  __shared__ unsigned short sm[36864];                     // 72 KB
  int tid = threadIdx.x, wave = tid >> 5, lane = tid & 31;
  int lid = blockIdx.x;
  int b = lid >> 10; int i0 = (lid >> 5) & 31, i1 = lid & 31;
  int base, stride;
  if (axis == 0){ base = ((b * 32) * 32 + i0) * 32 + i1; stride = 1024; }       // along D, (b,h,w)
  else if (axis == 1){ base = ((b * 32 + i0) * 32) * 32 + i1; stride = 32; }    // along H, (b,d,w)
  else { base = ((b * 32 + i0) * 32 + i1) * 32; stride = 1; }                   // along W, (b,d,h)

  // async DMA of the 32x128 bf16 line into LDS
  {
    const unsigned lds0 = lds_off_u32(sm);
    for (int c = tid; c < 512; c += 128){
      int tok = c >> 4;
      unsigned gb = (unsigned)((base + tok * stride) * 256 + (c & 15) * 16);
      async_copy_b128(lds0 + (unsigned)c * 16u, xmod, gb);
    }
    wait_async();
  }
  __syncthreads();

  // ---- QKV GEMM: (32x128) @ (128x384); wave handles 6 N-tiles ----
  v8f acc[2][6];
#pragma unroll
  for (int mt = 0; mt < 2; ++mt)
#pragma unroll
    for (int j = 0; j < 6; ++j) acc[mt][j] = (v8f){};
#pragma unroll
  for (int kt = 0; kt < 4; ++kt){
    v16bf a0 = ldsA(&sm[SM_XIN], 128, 0, kt * 32);
    v16bf a1 = ldsA(&sm[SM_XIN], 128, 16, kt * 32);
#pragma unroll
    for (int j = 0; j < 6; ++j){
      int nt = wave * 6 + j;
      v16bf bw = gfrag(qkvpack + (size_t)(kt * 24 + nt) * 512);
      acc[0][j] = wmma_bf16(a0, bw, acc[0][j]);
      acc[1][j] = wmma_bf16(a1, bw, acc[1][j]);
    }
  }
#pragma unroll
  for (int j = 0; j < 6; ++j){
    int n = (wave * 6 + j) * 16 + (lane & 15);
    float bs = qkv_b[n];
    int which = n >> 7, hh = (n >> 4) & 7, d = n & 15;
#pragma unroll
    for (int mt = 0; mt < 2; ++mt)
#pragma unroll
      for (int r = 0; r < 8; ++r){
        int m = mt * 16 + r + ((lane & 16) ? 8 : 0);
        sm[SM_QKV + ((which * 8 + hh) * 32 + m) * 16 + d] = f2bf(acc[mt][j][r] + bs);
      }
  }
  __syncthreads();

  // ---- rotate q,k (R_qk) and v (R_v); emit WMMA-operand layouts directly ----
  for (int i = tid; i < 8192; i += 128){                   // q: [h][tok][d(32, pad)] (A-layout)
    int hh = i >> 10, rem = i & 1023, tok = rem >> 5, d = rem & 31;
    float val = 0.f;
    if (d < 16){
      int g = d / 3, jj = d - g * 3;
      const float* R = &rotb[((axis * 8 + hh) * 2 + 0) * 9 + jj * 3];
      float s = 0.f;
#pragma unroll
      for (int ii = 0; ii < 3; ++ii){
        int src = g * 3 + ii;
        if (src < 16) s += bf2f(sm[SM_QKV + ((0 * 8 + hh) * 32 + tok) * 16 + src]) * R[ii];
      }
      val = s;
    }
    sm[SM_Q + (hh * 32 + tok) * 32 + d] = f2bf(val);
  }
  for (int i = tid; i < 8192; i += 128){                   // k^T: fragment-packed [h][nt][lane][16]
    int hh = i >> 10, rem = i & 1023, d = rem >> 5, tok = rem & 31;   // k = d (padded), n = tok
    float val = 0.f;
    if (d < 16){
      int g = d / 3, jj = d - g * 3;
      const float* R = &rotb[((axis * 8 + hh) * 2 + 0) * 9 + jj * 3];
      float s = 0.f;
#pragma unroll
      for (int ii = 0; ii < 3; ++ii){
        int src = g * 3 + ii;
        if (src < 16) s += bf2f(sm[SM_QKV + ((1 * 8 + hh) * 32 + tok) * 16 + src]) * R[ii];
      }
      val = s;
    }
    int nt = tok >> 4, ln, jf;
    bpos(d, tok & 15, ln, jf);
    sm[SM_KP + ((hh * 2 + nt) * 32 + ln) * 16 + jf] = f2bf(val);
  }
  for (int i = tid; i < 4096; i += 128){                   // v: fragment-packed [h][lane][16]; k=tok, n=d
    int hh = i >> 9, rem = i & 511, tok = rem >> 4, d = rem & 15;
    int g = d / 3, jj = d - g * 3;
    const float* R = &rotb[((axis * 8 + hh) * 2 + 1) * 9 + jj * 3];
    float s = 0.f;
#pragma unroll
    for (int ii = 0; ii < 3; ++ii){
      int src = g * 3 + ii;
      if (src < 16) s += bf2f(sm[SM_QKV + ((2 * 8 + hh) * 32 + tok) * 16 + src]) * R[ii];
    }
    int ln, jf;
    bpos(tok, d, ln, jf);
    sm[SM_VP + (hh * 32 + ln) * 16 + jf] = f2bf(s);
  }
  __syncthreads();

  // ---- attention per head (2 heads per wave) ----
  float wc = pbw[comp];
  for (int hi = 0; hi < 2; ++hi){
    int hh = wave * 2 + hi;
    const unsigned short* qh = &sm[SM_Q + hh * 1024];
    unsigned short* P = &sm[SM_P + hh * 1024];
    v16bf aq0 = ldsA(qh, 32, 0, 0),  aq1 = ldsA(qh, 32, 16, 0);
    v16bf bk0 = gfrag(&sm[SM_KP + (hh * 2 + 0) * 512]);
    v16bf bk1 = gfrag(&sm[SM_KP + (hh * 2 + 1) * 512]);
    v8f z = {};
    v8f s[2][2];
    s[0][0] = wmma_bf16(aq0, bk0, z); s[0][1] = wmma_bf16(aq0, bk1, z);
    s[1][0] = wmma_bf16(aq1, bk0, z); s[1][1] = wmma_bf16(aq1, bk1, z);
    // only the key-side positional bias survives the softmax: -lin[k]*w[comp]
    float add0 = -(-1.f + (float)(lane & 15) * (2.f / 31.f)) * wc;
    float add1 = -(-1.f + (float)((lane & 15) + 16) * (2.f / 31.f)) * wc;
#pragma unroll
    for (int mt = 0; mt < 2; ++mt){
#pragma unroll
      for (int r = 0; r < 8; ++r){
        float a0 = s[mt][0][r] * 0.25f + add0;              // scale = hd^-0.5
        float a1 = s[mt][1][r] * 0.25f + add1;
        float mx = fmaxf(a0, a1);
#pragma unroll
        for (int o = 1; o < 16; o <<= 1) mx = fmaxf(mx, __shfl_xor(mx, o, 32));
        float e0 = __expf(a0 - mx), e1 = __expf(a1 - mx);
        float ss = e0 + e1;
#pragma unroll
        for (int o = 1; o < 16; o <<= 1) ss += __shfl_xor(ss, o, 32);
        float inv = 1.f / ss;
        int m = mt * 16 + r + ((lane & 16) ? 8 : 0);
        P[m * 32 + (lane & 15)]      = f2bf(e0 * inv);
        P[m * 32 + 16 + (lane & 15)] = f2bf(e1 * inv);
      }
    }
    v16bf ap0 = ldsA(P, 32, 0, 0), ap1 = ldsA(P, 32, 16, 0);
    v16bf bv = gfrag(&sm[SM_VP + hh * 512]);
    v8f o0 = wmma_bf16(ap0, bv, z), o1 = wmma_bf16(ap1, bv, z);
    int c = hh * 16 + (lane & 15);
#pragma unroll
    for (int r = 0; r < 8; ++r){
      int m = r + ((lane & 16) ? 8 : 0);
      atomicAdd(&outacc[(size_t)(base + m * stride) * 128 + c], o0[r]);
      atomicAdd(&outacc[(size_t)(base + (16 + m) * stride) * 128 + c], o1[r]);
    }
  }
}

// ---------------- host ----------------
extern "C" void kernel_launch(void* const* d_in, const int* in_sizes, int n_in,
                              void* d_out, int out_size, void* d_ws, size_t ws_size,
                              hipStream_t stream){
  (void)in_sizes; (void)n_in; (void)out_size; (void)ws_size;
  const float* x       = (const float*)d_in[0];
  const float* pos_emb = (const float*)d_in[1];
  const float* qkv_w   = (const float*)d_in[2];
  const float* qkv_b   = (const float*)d_in[3];
  const float* lp_w1   = (const float*)d_in[4];
  const float* lp_b1   = (const float*)d_in[5];
  const float* lp_w2   = (const float*)d_in[6];
  const float* lp_b2   = (const float*)d_in[7];
  const float* mod_w1  = (const float*)d_in[8];
  const float* mod_b1  = (const float*)d_in[9];
  const float* mod_w2  = (const float*)d_in[10];
  const float* mod_b2  = (const float*)d_in[11];
  // pa_w/pa_b (12,13) intentionally unused: per-query softmax constants cancel exactly
  const float* proj_w  = (const float*)d_in[14];
  const float* proj_b  = (const float*)d_in[15];
  const float* pbw     = (const float*)d_in[16];
  const float* Ad      = (const float*)d_in[18];
  const float* Ah      = (const float*)d_in[19];
  const float* Aw      = (const float*)d_in[20];
  const float* Rd      = (const float*)d_in[21];
  const float* Rh      = (const float*)d_in[22];
  const float* Rw      = (const float*)d_in[23];

  char* ws = (char*)d_ws;
  size_t off = 0;
  auto take = [&](size_t n) -> void* { void* p = ws + off; off = (off + n + 255) & ~(size_t)255; return p; };
  unsigned short* pe_cl   = (unsigned short*)take(16777216);   // [65536][128] bf16 (later reused for x_mod)
  unsigned short* w1pack  = (unsigned short*)take(884736);
  unsigned short* qkvpack = (unsigned short*)take(98304);
  unsigned short* wp_lp2  = (unsigned short*)take(65536);
  unsigned short* wp_m1   = (unsigned short*)take(65536);
  unsigned short* wp_m2   = (unsigned short*)take(65536);
  unsigned short* wp_pr   = (unsigned short*)take(65536);
  float*          bufA    = (float*)take(33554432);            // lf_raw, reused for m1
  unsigned short* lf_bf   = (unsigned short*)take(16777216);
  float*          outacc  = (float*)take(33554432);
  float*          st1     = (float*)take(2048);
  float*          st2     = (float*)take(2048);
  float*          rotb    = (float*)take(1728);
  unsigned short* xmod    = pe_cl;                             // safe reuse: pe dead after conv

  // pack / setup
  pack_pe_kernel<<<32768, 256, 0, stream>>>(pos_emb, pe_cl);
  pack_wconv_kernel<<<1728, 256, 0, stream>>>(lp_w1, w1pack);
  pack_wtok_kernel<<<192, 256, 0, stream>>>(qkv_w, 24, qkvpack);
  pack_wtok_kernel<<<64, 256, 0, stream>>>(lp_w2, 8, wp_lp2);
  pack_wtok_kernel<<<64, 256, 0, stream>>>(mod_w1, 8, wp_m1);
  pack_wtok_kernel<<<64, 256, 0, stream>>>(mod_w2, 8, wp_m2);
  pack_wtok_kernel<<<64, 256, 0, stream>>>(proj_w, 8, wp_pr);
  rot_kernel<<<1, 32, 0, stream>>>(Ad, Ah, Aw, Rd, Rh, Rw, rotb);

  // modulation path
  conv3_kernel<<<2048, 256, 0, stream>>>(pe_cl, w1pack, lp_b1, bufA);
  stats_kernel<<<256, 256, 0, stream>>>(bufA, st1);
  token_gemm_kernel<<<512, 256, 0, stream>>>(bufA, 1, st1, 1, wp_lp2, lp_b2, 1, lf_bf, nullptr);
  token_gemm_kernel<<<512, 256, 0, stream>>>(lf_bf, 0, nullptr, 0, wp_m1, mod_b1, 0, bufA, nullptr);
  stats_kernel<<<256, 256, 0, stream>>>(bufA, st2);
  token_gemm_kernel<<<512, 256, 0, stream>>>(bufA, 1, st2, 1, wp_m2, mod_b2, 2, xmod, x);

  // axial attention (3 axes) accumulated into outacc
  hipMemsetAsync(outacc, 0, 33554432, stream);
  axial_attn_kernel<<<2048, 128, 0, stream>>>(xmod, qkvpack, qkv_b, rotb, pbw, 0, 2, outacc);
  axial_attn_kernel<<<2048, 128, 0, stream>>>(xmod, qkvpack, qkv_b, rotb, pbw, 1, 1, outacc);
  axial_attn_kernel<<<2048, 128, 0, stream>>>(xmod, qkvpack, qkv_b, rotb, pbw, 2, 0, outacc);

  // output projection -> NCDHW f32
  token_gemm_kernel<<<512, 256, 0, stream>>>(outacc, 1, nullptr, 0, wp_pr, proj_b, 3, d_out, nullptr);
}
